// DefenseTrajectoryPredictorLSTM_60258391163314
// MI455X (gfx1250) — matrix operand order repaired and checked by reading, not compiled, not run on hardware
//
#include <hip/hip_runtime.h>

typedef __attribute__((ext_vector_type(16))) _Float16 v16h;
typedef __attribute__((ext_vector_type(8)))  float    v8f;

#define ROWS 16      // batch rows per workgroup
#define HDIM 256     // hidden size
#define GDIM 1024    // 4*H gates
#define LDH  264     // padded f16 LDS stride (bank-conflict avoidance)

__device__ __forceinline__ float sigmoidf_(float x) {
  return 1.f / (1.f + __expf(-x));
}
__device__ __forceinline__ float tanhf_(float x) {
  // stable tanh via sigmoid: saturates correctly for large |x|
  return 2.f * sigmoidf_(2.f * x) - 1.f;
}

// A-fragment (16x32 f16) load per CDNA5 ISA layout:
// lanes 0-15: row M=lane, K in {0..7, 16..23}; lanes 16-31: row M=lane-16,
// K in {8..15, 24..31}. Two contiguous 16-byte LDS reads per lane.
__device__ __forceinline__ v16h load_a_frag(const _Float16* base, int ldk,
                                            int k0, int lm, int grp) {
  const _Float16* p = base + lm * ldk + k0 + grp * 8;
  v16h a;
#pragma unroll
  for (int e = 0; e < 8; ++e) a[e] = p[e];
#pragma unroll
  for (int e = 0; e < 8; ++e) a[e + 8] = p[e + 16];
  return a;
}

__device__ __forceinline__ v8f wmma_f16(v16h a, v16h b, v8f c) {
  return __builtin_amdgcn_wmma_f32_16x16x32_f16(false, a, false, b, (short)0, c,
                                                false, false);
}

// 16 x 64 tile accumulation: 4 N-tiles per pass to keep VGPR pressure low.
// K-loop deliberately NOT unrolled so B loads stay just-in-time (no spills).
__device__ __forceinline__ void gemm4(v8f* acc, const _Float16* alds, int aldk,
                                      int K, const _Float16* __restrict__ W,
                                      int ldw, int ncol0, int lane, int lm,
                                      int grp) {
#pragma unroll 1
  for (int k0 = 0; k0 < K; k0 += 32) {
    v16h a = load_a_frag(alds, aldk, k0, lm, grp);
    const _Float16* wrow = W + (size_t)(k0 + grp * 8) * ldw + ncol0 + lm;
#pragma unroll
    for (int n = 0; n < 4; ++n) {
      v16h b;
#pragma unroll
      for (int e = 0; e < 8; ++e) {
        b[e] = wrow[(size_t)e * ldw + n * 16];
        b[e + 8] = wrow[(size_t)(e + 16) * ldw + n * 16];
      }
      acc[n] = wmma_f16(a, b, acc[n]);
    }
  }
}

// ---------------------------------------------------------------------------
// Prep kernels: weight transpose+convert (K-major f16), bias fuse, input cvt
// ---------------------------------------------------------------------------
__global__ void transpose_to_f16(const float* __restrict__ src,
                                 _Float16* __restrict__ dst,
                                 int R, int C, int Cpad) {
  // src [R][C] row-major -> dst [Cpad][R] with zero padding rows >= C
  int total = Cpad * R;
  for (int idx = blockIdx.x * blockDim.x + threadIdx.x; idx < total;
       idx += blockDim.x * gridDim.x) {
    int c = idx / R, r = idx - c * R;
    float v = (c < C) ? src[(size_t)r * C + c] : 0.f;
    dst[(size_t)c * R + r] = (_Float16)v;
  }
}

__global__ void add_bias(const float* __restrict__ a, const float* __restrict__ b,
                         float* __restrict__ o, int n) {
  int i = blockIdx.x * blockDim.x + threadIdx.x;
  if (i < n) o[i] = a[i] + b[i];
}

__global__ void prep_x0(const float* __restrict__ cond /*[1024][200][158]*/,
                        _Float16* __restrict__ x0 /*[200][1024][160]*/) {
  const int total = 200 * 1024 * 160;
  for (int idx = blockIdx.x * blockDim.x + threadIdx.x; idx < total;
       idx += blockDim.x * gridDim.x) {
    int t = idx / (1024 * 160);
    int rem = idx - t * (1024 * 160);
    int b = rem / 160;
    int k = rem - b * 160;
    float v = (k < 158) ? cond[((size_t)b * 200 + t) * 158 + k] : 0.f;
    x0[idx] = (_Float16)v;
  }
}

// ---------------------------------------------------------------------------
// One LSTM layer over the full sequence. 64 workgroups x 16 batch rows each;
// h/c persist in LDS across the T-step loop; both GEMMs done with WMMA.
// ---------------------------------------------------------------------------
__global__ __launch_bounds__(256) void lstm_layer_kernel(
    const _Float16* __restrict__ xin, int Kin, int xld,  // [T][1024][xld] f16
    const _Float16* __restrict__ WihT,                   // [Kin][1024] f16
    const _Float16* __restrict__ WhhT,                   // [256][1024] f16
    const float* __restrict__ bias,                      // [1024] (bih+bhh)
    _Float16* __restrict__ ys,                           // [T][1024][256] f16
    float* __restrict__ hT, float* __restrict__ cT,      // [1024][256] f32
    int T) {
  __shared__ _Float16 sh_h[ROWS][LDH];
  __shared__ _Float16 sh_x[ROWS][LDH];
  __shared__ float    sh_c[ROWS][HDIM];
  __shared__ float    sh_g[ROWS][GDIM];

  const int b0 = blockIdx.x * ROWS;
  const int tid = threadIdx.x;
  const int wave = tid >> 5;
  const int lane = tid & 31;
  const int lm = lane & 15;
  const int grp = lane >> 4;

  for (int j = tid; j < ROWS * HDIM; j += 256) {
    int row = j >> 8, col = j & 255;
    sh_h[row][col] = (_Float16)0.f;
    sh_c[row][col] = 0.f;
  }
  __syncthreads();

  const int nw = Kin >> 1;  // dwords per input row
  const unsigned* xw_all = (const unsigned*)xin;
  unsigned* sxw = (unsigned*)&sh_x[0][0];

#pragma unroll 1
  for (int t = 0; t < T; ++t) {
    // stage x[t] batch tile into LDS (dword copies)
    const unsigned* xw = xw_all + ((size_t)t * 1024 + b0) * (xld >> 1);
    for (int j = tid; j < ROWS * nw; j += 256) {
      int row = j / nw, w = j - row * nw;
      sxw[row * (LDH >> 1) + w] = xw[(size_t)row * (xld >> 1) + w];
    }
    __syncthreads();

    // two passes of 4 N-tiles each (gate-column halves) -> low VGPR pressure
#pragma unroll 1
    for (int half = 0; half < 2; ++half) {
      v8f zero = {};
      v8f acc[4];
#pragma unroll
      for (int n = 0; n < 4; ++n) acc[n] = zero;
      const int ncol0 = wave * 128 + half * 64;
      gemm4(acc, &sh_h[0][0], LDH, HDIM, WhhT, GDIM, ncol0, lane, lm, grp);
      gemm4(acc, &sh_x[0][0], LDH, Kin,  WihT, GDIM, ncol0, lane, lm, grp);
#pragma unroll
      for (int n = 0; n < 4; ++n) {
        int col = ncol0 + n * 16 + lm;
#pragma unroll
        for (int r = 0; r < 8; ++r) sh_g[r + 8 * grp][col] = acc[n][r];
      }
    }
    __syncthreads();

    // elementwise LSTM cell (gate order i,f,g,o)
    for (int j = tid; j < ROWS * HDIM; j += 256) {
      int row = j >> 8, col = j & 255;
      float gi = sh_g[row][col]            + bias[col];
      float gf = sh_g[row][HDIM + col]     + bias[HDIM + col];
      float gg = sh_g[row][2 * HDIM + col] + bias[2 * HDIM + col];
      float go = sh_g[row][3 * HDIM + col] + bias[3 * HDIM + col];
      float c = sigmoidf_(gf) * sh_c[row][col] + sigmoidf_(gi) * tanhf_(gg);
      float h = sigmoidf_(go) * tanhf_(c);
      sh_c[row][col] = c;
      sh_h[row][col] = (_Float16)h;
      ys[(size_t)t * 1024 * HDIM + (size_t)(b0 + row) * HDIM + col] =
          (_Float16)h;
    }
    __syncthreads();
  }

  for (int j = tid; j < ROWS * HDIM; j += 256) {
    int row = j >> 8, col = j & 255;
    hT[(size_t)(b0 + row) * HDIM + col] = (float)sh_h[row][col];
    cT[(size_t)(b0 + row) * HDIM + col] = sh_c[row][col];
  }
}

// ---------------------------------------------------------------------------
// Autoregressive decoder: 3 LSTM cells + MLP head per step, batch-tile
// persistent in LDS. One launch; 125 steps.
// ---------------------------------------------------------------------------
__global__ __launch_bounds__(256) void decoder_kernel(
    const _Float16* __restrict__ WihT0,   // [32][1024]  (22 padded to 32)
    const _Float16* __restrict__ WihT12,  // [2][256][1024]
    const _Float16* __restrict__ WhhT,    // [3][256][1024]
    const float* __restrict__ bias,       // [3][1024]
    const float* __restrict__ h0,         // [3][1024][256]
    const float* __restrict__ c0,         // [3][1024][256]
    const _Float16* __restrict__ pW1T,    // [256][64]
    const float* __restrict__ pb1,        // [64]
    const float* __restrict__ pW2,        // [22][64] f32
    const float* __restrict__ pb2,        // [22]
    float* __restrict__ out) {            // [1024][125][22]
  __shared__ _Float16 sh_h[3][ROWS][LDH];
  __shared__ float    sh_c[3][ROWS][HDIM];
  __shared__ float    sh_g[ROWS][GDIM];
  __shared__ _Float16 sh_x[ROWS][40];     // padded feedback input (22 -> 32)
  __shared__ _Float16 sh_p1[ROWS][64];

  const int b0 = blockIdx.x * ROWS;
  const int tid = threadIdx.x;
  const int wave = tid >> 5;
  const int lane = tid & 31;
  const int lm = lane & 15;
  const int grp = lane >> 4;

  for (int l = 0; l < 3; ++l)
    for (int j = tid; j < ROWS * HDIM; j += 256) {
      int row = j >> 8, col = j & 255;
      size_t src = (size_t)l * 1024 * HDIM + (size_t)(b0 + row) * HDIM + col;
      sh_h[l][row][col] = (_Float16)h0[src];
      sh_c[l][row][col] = c0[src];
    }
  for (int j = tid; j < ROWS * 40; j += 256)
    sh_x[j / 40][j - (j / 40) * 40] = (_Float16)0.f;
  __syncthreads();

#pragma unroll 1
  for (int t = 0; t < 125; ++t) {
#pragma unroll 1
    for (int l = 0; l < 3; ++l) {
      const _Float16* Wih =
          (l == 0) ? WihT0 : (WihT12 + (size_t)(l - 1) * HDIM * GDIM);
      const int Kin = (l == 0) ? 32 : HDIM;
      const _Float16* xbase = (l == 0) ? &sh_x[0][0] : &sh_h[l - 1][0][0];
      const int xldk = (l == 0) ? 40 : LDH;
      const _Float16* Whh = WhhT + (size_t)l * HDIM * GDIM;

#pragma unroll 1
      for (int half = 0; half < 2; ++half) {
        v8f zero = {};
        v8f acc[4];
#pragma unroll
        for (int n = 0; n < 4; ++n) acc[n] = zero;
        const int ncol0 = wave * 128 + half * 64;
        gemm4(acc, &sh_h[l][0][0], LDH, HDIM, Whh, GDIM, ncol0, lane, lm, grp);
        gemm4(acc, xbase, xldk, Kin, Wih, GDIM, ncol0, lane, lm, grp);
#pragma unroll
        for (int n = 0; n < 4; ++n) {
          int col = ncol0 + n * 16 + lm;
#pragma unroll
          for (int r = 0; r < 8; ++r) sh_g[r + 8 * grp][col] = acc[n][r];
        }
      }
      __syncthreads();

      const float* bl = bias + l * GDIM;
      for (int j = tid; j < ROWS * HDIM; j += 256) {
        int row = j >> 8, col = j & 255;
        float gi = sh_g[row][col]            + bl[col];
        float gf = sh_g[row][HDIM + col]     + bl[HDIM + col];
        float gg = sh_g[row][2 * HDIM + col] + bl[2 * HDIM + col];
        float go = sh_g[row][3 * HDIM + col] + bl[3 * HDIM + col];
        float c = sigmoidf_(gf) * sh_c[l][row][col] + sigmoidf_(gi) * tanhf_(gg);
        float h = sigmoidf_(go) * tanhf_(c);
        sh_c[l][row][col] = c;
        sh_h[l][row][col] = (_Float16)h;
      }
      __syncthreads();
    }

    // p1 = relu(h2 @ pW1^T + pb1): 16x64, waves 0..3 take one 16-col tile each
    if (wave < 4) {
      v8f acc = {};
#pragma unroll 1
      for (int k0 = 0; k0 < HDIM; k0 += 32) {
        v16h a = load_a_frag(&sh_h[2][0][0], LDH, k0, lm, grp);
        v16h b;
#pragma unroll
        for (int e = 0; e < 8; ++e) {
          b[e] = pW1T[(size_t)(k0 + grp * 8 + e) * 64 + wave * 16 + lm];
          b[e + 8] = pW1T[(size_t)(k0 + 16 + grp * 8 + e) * 64 + wave * 16 + lm];
        }
        acc = wmma_f16(a, b, acc);
      }
#pragma unroll
      for (int r = 0; r < 8; ++r) {
        int row = r + 8 * grp, col = wave * 16 + lm;
        float v = acc[r] + pb1[col];
        sh_p1[row][col] = (_Float16)(v > 0.f ? v : 0.f);
      }
    }
    __syncthreads();

    // pred = p1 @ pW2^T + pb2 (tiny 16x22x64: plain VALU), write + feedback
    for (int j = tid; j < ROWS * 22; j += 256) {
      int row = j / 22, o = j - row * 22;
      float s = pb2[o];
#pragma unroll 8
      for (int p = 0; p < 64; ++p) s += (float)sh_p1[row][p] * pW2[o * 64 + p];
      out[(size_t)(b0 + row) * 125 * 22 + (size_t)t * 22 + o] = s;
      sh_x[row][o] = (_Float16)s;  // cols 22..31 stay zero from init
    }
    __syncthreads();
  }
}

// ---------------------------------------------------------------------------
// Host-side orchestration
// ---------------------------------------------------------------------------
extern "C" void kernel_launch(void* const* d_in, const int* in_sizes, int n_in,
                              void* d_out, int out_size, void* d_ws,
                              size_t ws_size, hipStream_t stream) {
  (void)in_sizes; (void)n_in; (void)out_size; (void)ws_size;
  const float* condition = (const float*)d_in[0];
  const float* enc_Wih0  = (const float*)d_in[1];
  const float* enc_WihR  = (const float*)d_in[2];
  const float* enc_Whh   = (const float*)d_in[3];
  const float* enc_bih   = (const float*)d_in[4];
  const float* enc_bhh   = (const float*)d_in[5];
  const float* dec_Wih0  = (const float*)d_in[6];
  const float* dec_WihR  = (const float*)d_in[7];
  const float* dec_Whh   = (const float*)d_in[8];
  const float* dec_bih   = (const float*)d_in[9];
  const float* dec_bhh   = (const float*)d_in[10];
  const float* pW1 = (const float*)d_in[11];
  const float* pb1 = (const float*)d_in[12];
  const float* pW2 = (const float*)d_in[13];
  const float* pb2 = (const float*)d_in[14];
  float* out = (float*)d_out;

  char* ws = (char*)d_ws;
  size_t off = 0;
  auto alloc = [&](size_t bytes) -> void* {
    void* p = ws + off;
    off += (bytes + 255) & ~(size_t)255;
    return p;
  };

  _Float16* X0   = (_Float16*)alloc((size_t)200 * 1024 * 160 * 2);
  _Float16* YSa  = (_Float16*)alloc((size_t)200 * 1024 * 256 * 2);
  _Float16* YSb  = (_Float16*)alloc((size_t)200 * 1024 * 256 * 2);
  _Float16* eWihT0  = (_Float16*)alloc((size_t)160 * 1024 * 2);
  _Float16* eWihT12 = (_Float16*)alloc((size_t)2 * 256 * 1024 * 2);
  _Float16* eWhhT   = (_Float16*)alloc((size_t)3 * 256 * 1024 * 2);
  _Float16* dWihT0  = (_Float16*)alloc((size_t)32 * 1024 * 2);
  _Float16* dWihT12 = (_Float16*)alloc((size_t)2 * 256 * 1024 * 2);
  _Float16* dWhhT   = (_Float16*)alloc((size_t)3 * 256 * 1024 * 2);
  _Float16* pW1T    = (_Float16*)alloc((size_t)256 * 64 * 2);
  float* biasE = (float*)alloc((size_t)3 * 1024 * 4);
  float* biasD = (float*)alloc((size_t)3 * 1024 * 4);
  float* h0    = (float*)alloc((size_t)3 * 1024 * 256 * 4);
  float* c0    = (float*)alloc((size_t)3 * 1024 * 256 * 4);

  // --- weight conversion (K-major f16) ---
  transpose_to_f16<<<256, 256, 0, stream>>>(enc_Wih0, eWihT0, 1024, 158, 160);
  for (int l = 0; l < 2; ++l) {
    transpose_to_f16<<<256, 256, 0, stream>>>(
        enc_WihR + (size_t)l * 1024 * 256, eWihT12 + (size_t)l * 256 * 1024,
        1024, 256, 256);
    transpose_to_f16<<<256, 256, 0, stream>>>(
        dec_WihR + (size_t)l * 1024 * 256, dWihT12 + (size_t)l * 256 * 1024,
        1024, 256, 256);
  }
  for (int l = 0; l < 3; ++l) {
    transpose_to_f16<<<256, 256, 0, stream>>>(
        enc_Whh + (size_t)l * 1024 * 256, eWhhT + (size_t)l * 256 * 1024,
        1024, 256, 256);
    transpose_to_f16<<<256, 256, 0, stream>>>(
        dec_Whh + (size_t)l * 1024 * 256, dWhhT + (size_t)l * 256 * 1024,
        1024, 256, 256);
  }
  transpose_to_f16<<<256, 256, 0, stream>>>(dec_Wih0, dWihT0, 1024, 22, 32);
  transpose_to_f16<<<64, 256, 0, stream>>>(pW1, pW1T, 64, 256, 256);
  add_bias<<<12, 256, 0, stream>>>(enc_bih, enc_bhh, biasE, 3 * 1024);
  add_bias<<<12, 256, 0, stream>>>(dec_bih, dec_bhh, biasD, 3 * 1024);
  prep_x0<<<2048, 256, 0, stream>>>(condition, X0);

  // --- encoder: 3 sequential layer kernels (64 WGs x 16 batch rows) ---
  lstm_layer_kernel<<<64, 256, 0, stream>>>(
      X0, 160, 160, eWihT0, eWhhT, biasE, YSa,
      h0, c0, 200);
  lstm_layer_kernel<<<64, 256, 0, stream>>>(
      YSa, 256, 256, eWihT12, eWhhT + (size_t)1 * 256 * 1024, biasE + 1024,
      YSb, h0 + (size_t)1 * 1024 * 256, c0 + (size_t)1 * 1024 * 256, 200);
  lstm_layer_kernel<<<64, 256, 0, stream>>>(
      YSb, 256, 256, eWihT12 + (size_t)256 * 1024,
      eWhhT + (size_t)2 * 256 * 1024, biasE + 2048,
      YSa, h0 + (size_t)2 * 1024 * 256, c0 + (size_t)2 * 1024 * 256, 200);

  // --- autoregressive decoder ---
  decoder_kernel<<<64, 256, 0, stream>>>(dWihT0, dWihT12, dWhhT, biasD, h0, c0,
                                         pW1T, pb1, pW2, pb2, out);
}
